// UserCoRelLogitWithAggre_47536698032841
// MI455X (gfx1250) — matrix operand order repaired and checked by reading, not compiled, not run on hardware
//
#include <hip/hip_runtime.h>
#include <hip/hip_bf16.h>
#include <math.h>

typedef __attribute__((ext_vector_type(16))) _Float16 v16h;
typedef __attribute__((ext_vector_type(8)))  _Float16 v8h;
typedef __attribute__((ext_vector_type(8)))  float    v8f;
typedef __attribute__((ext_vector_type(4)))  unsigned int v4u;
typedef __attribute__((ext_vector_type(8)))  int          v8i;
typedef __attribute__((ext_vector_type(4)))  int          v4i;

union V16 { v16h v; v8h h[2]; };

#define BZ      32
#define RVN     16
#define RVL     64
#define H       256
#define NTOK    1024            // RVN*RVL tokens per batch per side
#define HID     256
#define NEGV   -100000000.0f

// ---------------------------------------------------------------------------
// Kernel 1: per-token L2 normalize -> f16, plus fused word score for side A.
// One wave (32 lanes) per token; lane owns 8 contiguous channels.
// grid = (tokens/8, 2)  block = 256  (y==0 -> seq_a, y==1 -> seq_b)
// ---------------------------------------------------------------------------
__global__ __launch_bounds__(256) void normalize_f16_kernel(
    const float* __restrict__ seq_a, const float* __restrict__ seq_b,
    const int*   __restrict__ mask_a,
    const float* __restrict__ w_word, const float* __restrict__ b_word,
    _Float16* __restrict__ a16, _Float16* __restrict__ b16,
    float* __restrict__ word_out)
{
    const int wave = threadIdx.x >> 5;
    const int lane = threadIdx.x & 31;
    const int token = blockIdx.x * 8 + wave;          // [0, BZ*NTOK)
    const bool sideA = (blockIdx.y == 0);

    const float* row = (sideA ? seq_a : seq_b) + (size_t)token * H;
    float x[8];
    float ss = 0.0f;
#pragma unroll
    for (int i = 0; i < 8; ++i) {
        x[i] = row[lane * 8 + i];
        ss += x[i] * x[i];
    }
#pragma unroll
    for (int m = 16; m >= 1; m >>= 1) ss += __shfl_xor(ss, m);
    const float scale = 1.0f / (sqrtf(ss) + 1e-8f);

    v8h o;
#pragma unroll
    for (int i = 0; i < 8; ++i) o[i] = (_Float16)(x[i] * scale);
    _Float16* dst = (sideA ? a16 : b16) + (size_t)token * H + lane * 8;
    *(v8h*)dst = o;

    if (sideA) {
        float d = 0.0f;
#pragma unroll
        for (int i = 0; i < 8; ++i) d += x[i] * w_word[lane * 8 + i];
#pragma unroll
        for (int m = 16; m >= 1; m >>= 1) d += __shfl_xor(d, m);
        if (lane == 0) {
            const float ma = mask_a[token] ? 1.0f : 0.0f;
            word_out[token] = (d + b_word[0]) * ma;
        }
    }
}

// ---------------------------------------------------------------------------
// Kernel 2: batched GEMM (a_n @ b_n^T) fused with masked row-mean and row-max.
// grid = (64 M-tiles, 32 batches), block = 256 (8 waves).
// A tile (16x256 f16 = 8KB) staged to LDS once per block via the Tensor Data
// Mover; each wave builds register A fragments from LDS, then sweeps 8 of the
// 64 N-tiles issuing 8 x v_wmma_f32_16x16x32_f16 per tile with global B frags.
// ---------------------------------------------------------------------------
__global__ __launch_bounds__(256) void affinity_wmma_kernel(
    const _Float16* __restrict__ a16, const _Float16* __restrict__ b16,
    const int* __restrict__ mask_b, const float* __restrict__ word,
    float* __restrict__ feat)
{
    __shared__ _Float16 s_atile[16 * H];   // 8 KB A tile, row-major [16][256]
    __shared__ float s_mb[NTOK];
    __shared__ float s_red[256];
    __shared__ float s_sum[8 * 16];
    __shared__ float s_max[8 * 16];
    __shared__ float s_denom;

    const int b    = blockIdx.y;
    const int mt   = blockIdx.x;           // 16-row tile, 64 per batch
    const int tid  = threadIdx.x;
    const int wv   = tid >> 5;
    const int lane = tid & 31;
    const int half = lane >> 4;            // K-half selector (lane bit 4)
    const int l15  = lane & 15;            // row (A) / col (B,C) within tile

    // ---- TDM: async-copy the A tile into LDS (one descriptor, wave 0) ----
    if (wv == 0) {
        const unsigned long long ga =
            (unsigned long long)(uintptr_t)(a16 + ((size_t)b * NTOK + (size_t)mt * 16) * H);
        const unsigned int lds_off = (unsigned int)(uintptr_t)(&s_atile[0]);

        v4u g0;
        g0[0] = 1u;                                   // count=1, user descriptor
        g0[1] = lds_off;                              // LDS byte address
        g0[2] = (unsigned int)(ga & 0xFFFFFFFFu);     // global_addr[31:0]
        g0[3] = (unsigned int)((ga >> 32) & 0x01FFFFFFu) | (2u << 30); // type=2

        v8i g1;
        g1[0] = (int)(1u << 16);                      // data_size=1 (2 bytes)
        g1[1] = (int)((H & 0xFFFFu) << 16);           // tensor_dim0 = 256 (lo16)
        g1[2] = (int)((16u & 0xFFFFu) << 16);         // tensor_dim1 = 16
        g1[3] = (int)((H & 0xFFFFu) << 16);           // tile_dim0 = 256
        g1[4] = (int)(16u);                           // tile_dim1 = 16
        g1[5] = (int)(H);                             // tensor_dim0_stride = 256
        g1[6] = 0;
        g1[7] = 0;

        v4i gz4 = {0, 0, 0, 0};                       // 2-D tensor: groups 2/3 zero
        v8i gz8 = {0, 0, 0, 0, 0, 0, 0, 0};
        __builtin_amdgcn_tensor_load_to_lds(g0, g1, gz4, gz4, gz8, 0);
    }
    __builtin_amdgcn_s_wait_tensorcnt(0);

    // ---- mask_b to LDS (+ denominator for the masked mean) ----
    float p = 0.0f;
    for (int i = tid; i < NTOK; i += 256) {
        const float m = mask_b[b * NTOK + i] ? 1.0f : 0.0f;
        s_mb[i] = m;
        p += m;
    }
    s_red[tid] = p;
    __syncthreads();
    for (int s = 128; s > 0; s >>= 1) {
        if (tid < s) s_red[tid] += s_red[tid + s];
        __syncthreads();
    }
    if (tid == 0) s_denom = fmaxf(s_red[0], 1.0f);

    // ---- build A fragments from LDS (16x32 f16 layout: two 16B runs/lane) ----
    const _Float16* arow = &s_atile[l15 * H];
    v16h afrag[8];
#pragma unroll
    for (int kk = 0; kk < 8; ++kk) {
        V16 t;
        t.h[0] = *(const v8h*)(arow + kk * 32 + 8 * half);        // K = k0+8h..+7
        t.h[1] = *(const v8h*)(arow + kk * 32 + 16 + 8 * half);   // K = k0+16+8h..
        afrag[kk] = t.v;
    }

    float accs[8], accm[8];
#pragma unroll
    for (int r = 0; r < 8; ++r) { accs[r] = 0.0f; accm[r] = -3.0e38f; }

    const _Float16* bbase = b16 + (size_t)b * NTOK * H;

    // ---- sweep this wave's share of the 64 N-tiles ----
    for (int nt = wv; nt < 64; nt += 8) {
        const _Float16* brow = bbase + (size_t)(nt * 16 + l15) * H + 8 * half;
        __builtin_prefetch(brow + 8 * 16 * H, 0, 1);   // next tile (speculative)

        v8f c = {};
#pragma unroll
        for (int kk = 0; kk < 8; ++kk) {
            V16 t;                                     // B: K = k0+8h..+7, k0+16+8h..+7
            t.h[0] = *(const v8h*)(brow + kk * 32);
            t.h[1] = *(const v8h*)(brow + kk * 32 + 16);
            c = __builtin_amdgcn_wmma_f32_16x16x32_f16(
                    false, afrag[kk], false, t.v, (short)0, c, false, false);
        }
        const float mb = s_mb[nt * 16 + l15];          // this lane's column mask
#pragma unroll
        for (int r = 0; r < 8; ++r) {
            accs[r] += c[r] * mb;
            accm[r]  = fmaxf(accm[r], c[r]);
        }
    }

    // ---- reduce over the 16 columns held by each half-wave (xor 1,2,4,8) ----
#pragma unroll
    for (int m = 1; m < 16; m <<= 1) {
#pragma unroll
        for (int r = 0; r < 8; ++r) {
            accs[r] += __shfl_xor(accs[r], m);
            accm[r]  = fmaxf(accm[r], __shfl_xor(accm[r], m));
        }
    }
    if (l15 == 0) {
#pragma unroll
        for (int r = 0; r < 8; ++r) {
            const int ml = r + 8 * half;               // C layout: M = r + 8*half
            s_sum[wv * 16 + ml] = accs[r];
            s_max[wv * 16 + ml] = accm[r];
        }
    }
    __syncthreads();

    // ---- cross-wave combine, scale by word score, emit interleaved feat ----
    if (tid < 16) {
        float ssum = 0.0f, smax = -3.0e38f;
#pragma unroll
        for (int w = 0; w < 8; ++w) {
            ssum += s_sum[w * 16 + tid];
            smax  = fmaxf(smax, s_max[w * 16 + tid]);
        }
        const float mean = ssum / s_denom;
        const int gm = mt * 16 + tid;                  // global row in [0,1024)
        const float wvv = word[b * NTOK + gm];
        const int rv = gm >> 6, l = gm & 63;
        float* fo = feat + ((size_t)(b * RVN + rv)) * (2 * RVL);
        fo[2 * l]     = mean * wvv;
        fo[2 * l + 1] = smax * wvv;
    }
}

// ---------------------------------------------------------------------------
// Kernel 3: per (b, review) head: tanh MLP -> logit, masked softmax over 64,
// attention-weighted aggregation of seq_a. grid = 512, block = 256.
// ---------------------------------------------------------------------------
__global__ __launch_bounds__(256) void head_kernel(
    const float* __restrict__ feat, const float* __restrict__ word,
    const int* __restrict__ mask_a, const float* __restrict__ seq_a,
    const float* __restrict__ w1, const float* __restrict__ b1,
    const float* __restrict__ w2, const float* __restrict__ b2,
    float* __restrict__ out)
{
    __shared__ float s_feat[2 * RVL];
    __shared__ float s_red[256];
    __shared__ float s_attn[RVL];

    const int br  = blockIdx.x;            // b*16 + rv
    const int b   = br >> 4;
    const int rv  = br & 15;
    const int tid = threadIdx.x;

    if (tid < 2 * RVL) s_feat[tid] = feat[(size_t)br * (2 * RVL) + tid];
    __syncthreads();

    // hdd[k] = tanh(feat . w1[:,k] + b1[k]);  logit = hdd . w2 + b2
    float acc = b1[tid];
#pragma unroll 4
    for (int f = 0; f < 2 * RVL; ++f) acc += s_feat[f] * w1[f * HID + tid];
    const float hdd = tanhf(acc);
    s_red[tid] = hdd * w2[tid];
    __syncthreads();
    for (int s = 128; s > 0; s >>= 1) {
        if (tid < s) s_red[tid] += s_red[tid + s];
        __syncthreads();
    }
    if (tid == 0) out[br] = s_red[0] + b2[0];

    // masked softmax over the 64 word scores
    const int tok0 = br * RVL;
    if (tid < RVL) {
        const float wvv = word[b * NTOK + rv * RVL + tid];
        s_attn[tid] = mask_a[tok0 + tid] ? wvv : NEGV;     // TEMP == 1
    }
    __syncthreads();
    if (tid < RVL) s_red[tid] = s_attn[tid];
    __syncthreads();
    for (int s = 32; s > 0; s >>= 1) {
        if (tid < s) s_red[tid] = fmaxf(s_red[tid], s_red[tid + s]);
        __syncthreads();
    }
    const float mx = s_red[0];
    __syncthreads();
    if (tid < RVL) {
        const float e = __expf(s_attn[tid] - mx);
        s_attn[tid] = e;
        s_red[tid] = e;
    }
    __syncthreads();
    for (int s = 32; s > 0; s >>= 1) {
        if (tid < s) s_red[tid] += s_red[tid + s];
        __syncthreads();
    }
    const float inv = 1.0f / s_red[0];
    __syncthreads();
    if (tid < RVL) s_attn[tid] *= inv;
    __syncthreads();

    // aggre[h] = sum_l attn[l] * seq_a[b,rv,l,h]
    const float* sa = seq_a + (size_t)br * RVL * H;
    float a = 0.0f;
#pragma unroll 4
    for (int l = 0; l < RVL; ++l) a += s_attn[l] * sa[(size_t)l * H + tid];
    out[BZ * RVN + (size_t)br * H + tid] = a;
}

// ---------------------------------------------------------------------------
extern "C" void kernel_launch(void* const* d_in, const int* in_sizes, int n_in,
                              void* d_out, int out_size, void* d_ws, size_t ws_size,
                              hipStream_t stream) {
    const float* seq_a  = (const float*)d_in[0];
    const float* seq_b  = (const float*)d_in[1];
    const int*   mask_a = (const int*)  d_in[2];
    const int*   mask_b = (const int*)  d_in[3];
    const float* w_word = (const float*)d_in[4];
    const float* b_word = (const float*)d_in[5];
    const float* w1     = (const float*)d_in[6];
    const float* b1     = (const float*)d_in[7];
    const float* w2     = (const float*)d_in[8];
    const float* b2     = (const float*)d_in[9];
    float* out = (float*)d_out;

    char* ws = (char*)d_ws;
    const size_t SIDE_BYTES = (size_t)BZ * NTOK * H * sizeof(_Float16);  // 16 MB
    _Float16* a16  = (_Float16*)(ws);
    _Float16* b16  = (_Float16*)(ws + SIDE_BYTES);
    float*    word = (float*)   (ws + 2 * SIDE_BYTES);
    float*    feat = (float*)   (ws + 2 * SIDE_BYTES + (size_t)BZ * NTOK * sizeof(float));

    // K1: normalize both sides to f16 (+ word scores for side A)
    normalize_f16_kernel<<<dim3((BZ * NTOK) / 8, 2), 256, 0, stream>>>(
        seq_a, seq_b, mask_a, w_word, b_word, a16, b16, word);

    // K2: TDM-staged WMMA affinity GEMM fused with masked mean / max
    affinity_wmma_kernel<<<dim3(64, BZ), 256, 0, stream>>>(
        a16, b16, mask_b, word, feat);

    // K3: MLP head + masked softmax + aggregation
    head_kernel<<<BZ * RVN, 256, 0, stream>>>(
        feat, word, mask_a, seq_a, w1, b1, w2, b2, out);
}
